// CrossAttentionExpert_9466107920783
// MI455X (gfx1250) — matrix operand order repaired and checked by reading, not compiled, not run on hardware
//
#include <hip/hip_runtime.h>
#include <hip/hip_bf16.h>
#include <stdint.h>

// ---------------------------------------------------------------------------
// Types
// ---------------------------------------------------------------------------
typedef __bf16 bf16;
typedef __attribute__((ext_vector_type(16))) __bf16 v16bf;
typedef __attribute__((ext_vector_type(8)))  __bf16 v8bf;
typedef __attribute__((ext_vector_type(8)))  float  v8f;

#define B_ROWS 16384
#define E_DIM  1024
#define NHEAD  16
#define HDIM   64

#define BM 128
#define BN 128
#define BK 32

// ---------------------------------------------------------------------------
// f32 -> bf16 conversion (vectorized float4, n is a multiple of 1024)
// ---------------------------------------------------------------------------
__global__ __launch_bounds__(256) void cvt_f32_bf16(const float* __restrict__ in,
                                                    bf16* __restrict__ out,
                                                    long long n) {
  long long i = ((long long)blockIdx.x * blockDim.x + threadIdx.x) * 4;
  if (i + 3 < n) {
    float4 v = *(const float4*)(in + i);
    out[i + 0] = (bf16)v.x;
    out[i + 1] = (bf16)v.y;
    out[i + 2] = (bf16)v.z;
    out[i + 3] = (bf16)v.w;
  }
}

// ---------------------------------------------------------------------------
// Tiled GEMM: C[M,N] = A[M,K] @ B[N,K]^T + bias[N]
//   A, B in bf16 row-major (K contiguous for both -> "x @ w.T" layout).
//   Accumulate fp32 with v_wmma_f32_16x16x32_bf16.
//   Global->LDS via CDNA5 async-to-LDS, double buffered.
//   Output: bf16 (intermediates) or fp32 (final Y), row stride ldc.
// Block: 256 threads = 8 waves arranged 2 (M) x 4 (N); each wave: 64x32.
// ---------------------------------------------------------------------------
template <bool STORE_BF16>
__global__ __launch_bounds__(256) void gemm_bt_bf16(
    const bf16* __restrict__ A, const bf16* __restrict__ Bm,
    const float* __restrict__ bias,
    float* __restrict__ Cf, bf16* __restrict__ Cb,
    int M, int N, int K, int ldc) {
  __shared__ bf16 sA[2][BM * BK];
  __shared__ bf16 sB[2][BN * BK];

  const int tid  = threadIdx.x;
  const int lane = tid & 31;
  const int wave = tid >> 5;
  const int wm   = wave >> 2;          // 0..1  (64-row strip)
  const int wn   = wave & 3;           // 0..3  (32-col strip)
  const int m0   = blockIdx.y * BM;
  const int n0   = blockIdx.x * BN;
  const int half = lane >> 4;          // K-half selector (this code's contiguous k = 16*half + s)
  const int mrow = lane & 15;

  // Stage a (BM x BK) A tile and a (BN x BK) B tile into LDS asynchronously.
  // 512 16-byte chunks per matrix tile -> 2 chunks/thread/matrix = 4 async ops.
  auto issue = [&](int buf, int kt) {
#pragma unroll
    for (int i = 0; i < 2; ++i) {
      int c   = tid + i * 256;
      int row = c >> 2;                // tile row (0..127)
      int kc  = (c & 3) * 8;           // bf16 element offset inside row
      uint32_t la = (uint32_t)(uintptr_t)&sA[buf][row * BK + kc];
      const bf16* ga = A + (size_t)(m0 + row) * K + (size_t)kt * BK + kc;
      asm volatile("global_load_async_to_lds_b128 %0, %1, off"
                   :: "v"(la), "v"(ga) : "memory");
      uint32_t lb = (uint32_t)(uintptr_t)&sB[buf][row * BK + kc];
      const bf16* gb = Bm + (size_t)(n0 + row) * K + (size_t)kt * BK + kc;
      asm volatile("global_load_async_to_lds_b128 %0, %1, off"
                   :: "v"(lb), "v"(gb) : "memory");
    }
  };

  v8f acc[4][2] = {};

  const int nk = K / BK;
  issue(0, 0);
  for (int kt = 0; kt < nk; ++kt) {
    const int buf = kt & 1;
    if (kt + 1 < nk) {
      issue(buf ^ 1, kt + 1);
      // 4 newer async ops outstanding; wait until only they remain.
      asm volatile("s_wait_asynccnt 0x4" ::: "memory");
    } else {
      asm volatile("s_wait_asynccnt 0x0" ::: "memory");
    }
    __syncthreads();

    // A fragments (16x32): lane<16 holds K0-15, lane>=16 K16-31.
    v16bf afr[4];
#pragma unroll
    for (int ti = 0; ti < 4; ++ti) {
      const bf16* base = &sA[buf][(wm * 64 + ti * 16 + mrow) * BK];
      v8bf lo = *(const v8bf*)(base + half * 16);
      v8bf hi = *(const v8bf*)(base + half * 16 + 8);
      union { v16bf v; v8bf h[2]; } u;
      u.h[0] = lo; u.h[1] = hi;
      afr[ti] = u.v;
    }
    // B fragments (32x16): lane = N column, lane group selects K half.
    v16bf bfr[2];
#pragma unroll
    for (int tj = 0; tj < 2; ++tj) {
      const bf16* base = &sB[buf][(wn * 32 + tj * 16 + mrow) * BK];
      bfr[tj] = *(const v16bf*)(base + half * 16);
    }

#pragma unroll
    for (int ti = 0; ti < 4; ++ti)
#pragma unroll
      for (int tj = 0; tj < 2; ++tj)
        acc[ti][tj] = __builtin_amdgcn_wmma_f32_16x16x32_bf16(
            false, afr[ti], false, bfr[tj], (short)0, acc[ti][tj], false, false);

    __syncthreads();
  }

  // Epilogue: C layout per ISA (VGPR r -> M = r + 8*half, lane&15 -> N).
#pragma unroll
  for (int ti = 0; ti < 4; ++ti) {
#pragma unroll
    for (int tj = 0; tj < 2; ++tj) {
      const int col = n0 + wn * 32 + tj * 16 + mrow;
      const float bv = bias ? bias[col] : 0.0f;
#pragma unroll
      for (int r = 0; r < 8; ++r) {
        const int row = m0 + wm * 64 + ti * 16 + r + 8 * half;
        const float v = acc[ti][tj][r] + bv;
        if (STORE_BF16) Cb[(size_t)row * ldc + col] = (bf16)v;
        else            Cf[(size_t)row * ldc + col] = v;
      }
    }
  }
}

// ---------------------------------------------------------------------------
// Attention with 2 keys: one wave per head, lane covers 2 dims.
// Q:[B,E]  KV1,KV2:[B,2E] (k|v)  Out:[B,E]   all bf16.
// ---------------------------------------------------------------------------
__global__ __launch_bounds__(512) void attn2_kernel(
    const bf16* __restrict__ Q, const bf16* __restrict__ KV1,
    const bf16* __restrict__ KV2, bf16* __restrict__ Out) {
  const int b    = blockIdx.x;
  const int h    = threadIdx.x >> 5;
  const int lane = threadIdx.x & 31;
  const int d    = h * HDIM + lane * 2;

  const size_t qo  = (size_t)b * E_DIM + d;
  const size_t kvo = (size_t)b * (2 * E_DIM);

  const float q0 = (float)Q[qo],            q1 = (float)Q[qo + 1];
  const float k10 = (float)KV1[kvo + d],    k11 = (float)KV1[kvo + d + 1];
  const float k20 = (float)KV2[kvo + d],    k21 = (float)KV2[kvo + d + 1];
  const float v10 = (float)KV1[kvo + E_DIM + d], v11 = (float)KV1[kvo + E_DIM + d + 1];
  const float v20 = (float)KV2[kvo + E_DIM + d], v21 = (float)KV2[kvo + E_DIM + d + 1];

  float s1 = q0 * k10 + q1 * k11;
  float s2 = q0 * k20 + q1 * k21;
#pragma unroll
  for (int off = 16; off > 0; off >>= 1) {
    s1 += __shfl_xor(s1, off, 32);
    s2 += __shfl_xor(s2, off, 32);
  }
  s1 *= 0.125f;  // 1/sqrt(64)
  s2 *= 0.125f;
  const float mx = fmaxf(s1, s2);
  const float e1 = __expf(s1 - mx), e2 = __expf(s2 - mx);
  const float inv = 1.0f / (e1 + e2);
  const float w1 = e1 * inv, w2 = e2 * inv;

  Out[qo]     = (bf16)(w1 * v10 + w2 * v20);
  Out[qo + 1] = (bf16)(w1 * v11 + w2 * v21);
}

// ---------------------------------------------------------------------------
// Row LayerNorm: one 256-thread block per row of Y[B,E] (fp32) -> d_out fp32.
// ---------------------------------------------------------------------------
__global__ __launch_bounds__(256) void ln_kernel(
    const float* __restrict__ Y, const float* __restrict__ g,
    const float* __restrict__ bta, float* __restrict__ out) {
  __shared__ float red[2][8];
  const int row = blockIdx.x;
  const float* y = Y + (size_t)row * E_DIM;

  float s = 0.0f, s2 = 0.0f, vals[4];
#pragma unroll
  for (int i = 0; i < 4; ++i) {
    float v = y[threadIdx.x + i * 256];
    vals[i] = v; s += v; s2 += v * v;
  }
#pragma unroll
  for (int off = 16; off > 0; off >>= 1) {
    s  += __shfl_xor(s,  off, 32);
    s2 += __shfl_xor(s2, off, 32);
  }
  const int wave = threadIdx.x >> 5, lane = threadIdx.x & 31;
  if (lane == 0) { red[0][wave] = s; red[1][wave] = s2; }
  __syncthreads();
  if (wave == 0) {
    float a = lane < 8 ? red[0][lane] : 0.0f;
    float b = lane < 8 ? red[1][lane] : 0.0f;
#pragma unroll
    for (int off = 4; off > 0; off >>= 1) {
      a += __shfl_xor(a, off, 32);
      b += __shfl_xor(b, off, 32);
    }
    if (lane == 0) { red[0][0] = a; red[1][0] = b; }
  }
  __syncthreads();
  const float mean = red[0][0] * (1.0f / E_DIM);
  const float var  = red[1][0] * (1.0f / E_DIM) - mean * mean;
  const float rstd = rsqrtf(var + 1e-5f);
  float* o = out + (size_t)row * E_DIM;
#pragma unroll
  for (int i = 0; i < 4; ++i) {
    const int c = threadIdx.x + i * 256;
    o[c] = (vals[i] - mean) * rstd * g[c] + bta[c];
  }
}

// ---------------------------------------------------------------------------
// Host-side orchestration
// ---------------------------------------------------------------------------
static inline void launch_cvt(const float* in, bf16* out, long long n, hipStream_t s) {
  cvt_f32_bf16<<<(int)(n / (256 * 4)), 256, 0, s>>>(in, out, n);
}

extern "C" void kernel_launch(void* const* d_in, const int* in_sizes, int n_in,
                              void* d_out, int out_size, void* d_ws, size_t ws_size,
                              hipStream_t stream) {
  (void)in_sizes; (void)n_in; (void)out_size; (void)ws_size;
  const int Bn = B_ROWS, E = E_DIM;

  // Inputs (setup_inputs order)
  const float* X[3]      = {(const float*)d_in[0], (const float*)d_in[1], (const float*)d_in[2]};
  const float* w_in[3]   = {(const float*)d_in[3], (const float*)d_in[7], (const float*)d_in[11]};
  const float* b_in[3]   = {(const float*)d_in[4], (const float*)d_in[8], (const float*)d_in[12]};
  const float* w_out[3]  = {(const float*)d_in[5], (const float*)d_in[9], (const float*)d_in[13]};
  const float* b_out[3]  = {(const float*)d_in[6], (const float*)d_in[10], (const float*)d_in[14]};
  const float* w_fuse    = (const float*)d_in[15];
  const float* b_fuse    = (const float*)d_in[16];
  const float* ln_g      = (const float*)d_in[17];
  const float* ln_b      = (const float*)d_in[18];

  // Workspace carve-out
  char* wsp = (char*)d_ws;
  size_t off = 0;
  auto carve = [&](size_t bytes) { void* p = wsp + off; off += bytes; return p; };
  const size_t BE2 = (size_t)Bn * E * 2;

  bf16* XB[3];    for (int s = 0; s < 3; ++s) XB[s]    = (bf16*)carve(BE2);
  bf16* WIN[3];   for (int s = 0; s < 3; ++s) WIN[s]   = (bf16*)carve((size_t)3 * E * E * 2);
  bf16* WOUT[3];  for (int s = 0; s < 3; ++s) WOUT[s]  = (bf16*)carve((size_t)E * E * 2);
  bf16* WFUSE    = (bf16*)carve((size_t)3 * E * E * 2);
  bf16* Qb[3];    for (int s = 0; s < 3; ++s) Qb[s]    = (bf16*)carve(BE2);
  bf16* KV[3][2]; for (int s = 0; s < 3; ++s) for (int j = 0; j < 2; ++j) KV[s][j] = (bf16*)carve(2 * BE2);
  bf16* ATT[3];   for (int s = 0; s < 3; ++s) ATT[s]   = (bf16*)carve(BE2);
  bf16* CONCAT   = (bf16*)carve(3 * BE2);
  float* Y       = (float*)carve((size_t)Bn * E * 4);

  // 1) convert inputs + weights to bf16
  for (int s = 0; s < 3; ++s) launch_cvt(X[s], XB[s], (long long)Bn * E, stream);
  for (int s = 0; s < 3; ++s) launch_cvt(w_in[s], WIN[s], 3LL * E * E, stream);
  for (int s = 0; s < 3; ++s) launch_cvt(w_out[s], WOUT[s], (long long)E * E, stream);
  launch_cvt(w_fuse, WFUSE, 3LL * E * E, stream);

  const dim3 blk(256);
  const int otherOf[3][2] = {{1, 2}, {0, 2}, {0, 1}};  // kv order per reference

  // 2) Q projections: Q[s] = X[s] @ Wq[s]^T + bq   (N = E)
  for (int s = 0; s < 3; ++s) {
    dim3 grid(E / BN, Bn / BM);
    gemm_bt_bf16<true><<<grid, blk, 0, stream>>>(
        XB[s], WIN[s], b_in[s], nullptr, Qb[s], Bn, E, E, E);
  }
  // 3) K|V projections: KV[s][j] = X[other] @ [Wk[s];Wv[s]]^T + [bk;bv]  (N = 2E)
  for (int s = 0; s < 3; ++s) {
    for (int j = 0; j < 2; ++j) {
      dim3 grid((2 * E) / BN, Bn / BM);
      gemm_bt_bf16<true><<<grid, blk, 0, stream>>>(
          XB[otherOf[s][j]], WIN[s] + (size_t)E * E, b_in[s] + E,
          nullptr, KV[s][j], Bn, 2 * E, E, 2 * E);
    }
  }
  // 4) attention (2 keys, per-head softmax)
  for (int s = 0; s < 3; ++s)
    attn2_kernel<<<Bn, 512, 0, stream>>>(Qb[s], KV[s][0], KV[s][1], ATT[s]);

  // 5) out-proj straight into the concat buffer at column offset s*E
  for (int s = 0; s < 3; ++s) {
    dim3 grid(E / BN, Bn / BM);
    gemm_bt_bf16<true><<<grid, blk, 0, stream>>>(
        ATT[s], WOUT[s], b_out[s], nullptr, CONCAT + (size_t)s * E, Bn, E, E, 3 * E);
  }
  // 6) fuse GEMM: Y = concat @ w_fuse^T + b_fuse  (K = 3E, fp32 out)
  {
    dim3 grid(E / BN, Bn / BM);
    gemm_bt_bf16<false><<<grid, blk, 0, stream>>>(
        CONCAT, WFUSE, b_fuse, Y, nullptr, Bn, E, 3 * E, E);
  }
  // 7) LayerNorm -> d_out (fp32)
  ln_kernel<<<Bn, 256, 0, stream>>>(Y, ln_g, ln_b, (float*)d_out);
}
